// DIN_14276471292370
// MI455X (gfx1250) — matrix operand order repaired and checked by reading, not compiled, not run on hardware
//
#include <hip/hip_runtime.h>
#include <hip/hip_bf16.h>

// Problem sizes (fixed by the reference).
#define BB      4096
#define SS      200
#define EE      128
#define ATT_H   128
#define MLP_H   256
#define S_PAD   208   // 13 * 16 tiles of S

typedef __attribute__((ext_vector_type(16))) __bf16 v16bf;
typedef __attribute__((ext_vector_type(8)))  float  v8f;

union FragBF { v16bf v; unsigned u[8]; };

static __device__ __forceinline__ unsigned short f32_bf16(float f) {
  unsigned u = __float_as_uint(f);
  u += 0x7FFFu + ((u >> 16) & 1u);      // round-to-nearest-even
  return (unsigned short)(u >> 16);
}
static __device__ __forceinline__ unsigned pack_bf16(float lo, float hi) {
  return (unsigned)f32_bf16(lo) | ((unsigned)f32_bf16(hi) << 16);
}
static __device__ __forceinline__ float bf16_f32(unsigned short h) {
  return __uint_as_float(((unsigned)h) << 16);
}

// A-fragment (16x32 bf16, MxK) holding the TRANSPOSE of a row-major f32
// weight W:  A[m][k] = W[k*ld + mBase + m], converted to bf16.
// ISA A layout: lanes 0-15 -> M=lane; r<4: K=2r,2r+1 ; r>=4: K=16+2(r-4);
//               lanes 16-31 add +8 to the K base.
static __device__ __forceinline__ v16bf load_fragAT_global(
    const float* __restrict__ W, int ld, int kBase, int mBase) {
  const int lane = threadIdx.x & 31;
  const int m  = lane & 15;
  const int hi = lane >> 4;
  FragBF f;
#pragma unroll
  for (int r = 0; r < 8; ++r) {
    const int k0 = kBase + ((r < 4) ? (hi * 8 + 2 * r) : (16 + hi * 8 + 2 * (r - 4)));
    f.u[r] = pack_bf16(W[(size_t)k0 * ld + mBase + m],
                       W[(size_t)(k0 + 1) * ld + mBase + m]);
  }
  return f.v;
}

// B-fragment (32x16 bf16, KxN) holding the TRANSPOSE of a row-major bf16
// LDS buffer:  B[k][n] = s[(nBase+n)*ld + kBase + k].
// K runs along the contiguous axis -> pure dword LDS loads (vectorizable).
// ISA B layout: lanes 0-15 -> N=lane; r<4: K=2r,2r+1 ; r>=4: K=16+2(r-4);
//               lanes 16-31 add +8 to the K base.
static __device__ __forceinline__ v16bf load_fragBT_lds(
    const unsigned short* s, int ld, int kBase, int nBase) {
  const int lane = threadIdx.x & 31;
  const int n  = lane & 15;
  const int hi = lane >> 4;
  const unsigned* row = (const unsigned*)(s + (size_t)(nBase + n) * ld);
  FragBF f;
#pragma unroll
  for (int r = 0; r < 8; ++r)
    f.u[r] = row[(kBase + ((r < 4) ? (hi * 8 + 2 * r) : (16 + hi * 8 + 2 * (r - 4)))) >> 1];
  return f.v;
}

// ---------------------------------------------------------------------------
// Kernel 1: per batch row. Computes h^T = W_b^T x behavior^T with bf16 WMMA,
// so D has lane = behavior row s, VGPR = h-column. The score dot
// score[s] = sum_i relu(h[s][i]+cproj[i]+b[i]) * v[i] then reduces over the
// VGPR axis (8 in-register FMAs) + one ds_add_f32 per lane. No shuffles.
// Then softmax + attention-weighted sum from the LDS bf16 behavior copy.
// ---------------------------------------------------------------------------
__global__ __launch_bounds__(256)
void din_attn_kernel(const float* __restrict__ beh,
                     const float* __restrict__ cand,
                     const float* __restrict__ fc1_w,
                     const float* __restrict__ fc1_b,
                     const float* __restrict__ fc2_w,
                     float* __restrict__ xws) {
  __shared__ __align__(16) unsigned short sBeh[S_PAD * EE];  // 53.25 KB bf16
  __shared__ float sCproj[ATT_H];
  __shared__ float sV[ATT_H];
  __shared__ float sScores[S_PAD];

  const int b    = blockIdx.x;
  const int tid  = threadIdx.x;
  const int wave = tid >> 5;
  const int lane = tid & 31;
  const int n    = lane & 15;
  const int hi   = lane >> 4;

  // Stage behavior row -> bf16 LDS with float4 global loads; zero-pad 200..207.
  const float* behB = beh + (size_t)b * SS * EE;
  {
    const float4* behV = (const float4*)behB;
    uint2* sBehV = (uint2*)sBeh;
    for (int i4 = tid; i4 < (S_PAD * EE) / 4; i4 += 256) {
      float4 v = make_float4(0.f, 0.f, 0.f, 0.f);
      if (i4 < (SS * EE) / 4) v = behV[i4];
      sBehV[i4] = make_uint2(pack_bf16(v.x, v.y), pack_bf16(v.z, v.w));
    }
  }
  if (tid < S_PAD) sScores[tid] = 0.0f;

  // Candidate projection through W_c (fc1_w rows 128..255) + bias, fc2 vector.
  const float* candB = cand + (size_t)b * EE;
  if (tid < ATT_H) {
    float acc = fc1_b[tid];
    for (int e = 0; e < EE; ++e)
      acc += candB[e] * fc1_w[(size_t)(EE + e) * ATT_H + tid];
    sCproj[tid] = acc;
    sV[tid] = fc2_w[tid];
  }
  __syncthreads();

  // Each wave owns a 16-wide slice of h-columns (the M axis of the WMMA).
  // A = W_b^T fragments for K = 0..127, kept in VGPRs for the whole loop.
  const int mBase = wave * 16;
  const v16bf aw0 = load_fragAT_global(fc1_w, ATT_H, 0,  mBase);
  const v16bf aw1 = load_fragAT_global(fc1_w, ATT_H, 32, mBase);
  const v16bf aw2 = load_fragAT_global(fc1_w, ATT_H, 64, mBase);
  const v16bf aw3 = load_fragAT_global(fc1_w, ATT_H, 96, mBase);

  // Per-VGPR-row constants: h-column index i = mBase + r + 8*hi.
  float cbr[8], vvr[8];
#pragma unroll
  for (int r = 0; r < 8; ++r) {
    const int i = mBase + r + hi * 8;
    cbr[r] = sCproj[i];
    vvr[r] = sV[i];
  }

  for (int mt = 0; mt < S_PAD / 16; ++mt) {
    const int sBase = mt * 16;
    // B = behavior^T fragments from LDS (contiguous-K dword loads).
    v16bf b0 = load_fragBT_lds(sBeh, EE, 0,  sBase);
    v16bf b1 = load_fragBT_lds(sBeh, EE, 32, sBase);
    v16bf b2 = load_fragBT_lds(sBeh, EE, 64, sBase);
    v16bf b3 = load_fragBT_lds(sBeh, EE, 96, sBase);
    v8f c = {};
    c = __builtin_amdgcn_wmma_f32_16x16x32_bf16(false, aw0, false, b0, (short)0, c, false, false);
    c = __builtin_amdgcn_wmma_f32_16x16x32_bf16(false, aw1, false, b1, (short)0, c, false, false);
    c = __builtin_amdgcn_wmma_f32_16x16x32_bf16(false, aw2, false, b2, (short)0, c, false, false);
    c = __builtin_amdgcn_wmma_f32_16x16x32_bf16(false, aw3, false, b3, (short)0, c, false, false);

    // D[r][n] = h[s = sBase+n][i = mBase+r+8hi]. Reduce over r in registers.
    float part = 0.0f;
#pragma unroll
    for (int r = 0; r < 8; ++r) {
      float h = c[r] + cbr[r];
      h = h > 0.0f ? h : 0.0f;
      part = fmaf(h, vvr[r], part);
    }
    atomicAdd(&sScores[sBase + n], part);   // 16 distinct addrs, 2-way halves
  }
  __syncthreads();

  // Softmax over the 200 valid scores (fc2_b is a constant shift: no-op).
  if (wave == 0) {
    float mx = -3.0e38f;
    for (int i = lane; i < SS; i += 32) mx = fmaxf(mx, sScores[i]);
#pragma unroll
    for (int o = 16; o > 0; o >>= 1) mx = fmaxf(mx, __shfl_xor(mx, o));
    float sum = 0.0f;
    for (int i = lane; i < SS; i += 32) {
      float e = __expf(sScores[i] - mx);
      sScores[i] = e;
      sum += e;
    }
#pragma unroll
    for (int o = 16; o > 0; o >>= 1) sum += __shfl_xor(sum, o);
    const float inv = 1.0f / sum;
    for (int i = lane; i < SS; i += 32) sScores[i] *= inv;
  }
  __syncthreads();

  // user_interest[e] = sum_s attn[s]*behavior[s][e] from the LDS bf16 copy
  // (behavior touches HBM exactly once). Emit x = [user_interest, cand].
  float* xrow = xws + (size_t)b * (2 * EE);
  if (tid < EE) {
    float acc = 0.0f;
    for (int s = 0; s < SS; ++s)
      acc += sScores[s] * bf16_f32(sBeh[s * EE + tid]);
    xrow[tid] = acc;
  } else {
    xrow[EE + (tid - EE)] = candB[tid - EE];
  }
}

// ---------------------------------------------------------------------------
// Kernel 2: MLP head, same transposed scheme. z^T = W1^T x x^T; the logit
// dot z . mlp2_w reduces over the VGPR axis, one ds_add_f32 per lane.
// ---------------------------------------------------------------------------
__global__ __launch_bounds__(128)
void din_mlp_kernel(const float* __restrict__ xws,
                    const float* __restrict__ mlp1_w,
                    const float* __restrict__ mlp1_b,
                    const float* __restrict__ mlp2_w,
                    const float* __restrict__ mlp2_b,
                    float* __restrict__ out) {
  __shared__ __align__(16) unsigned short sX[16 * (2 * EE)];  // 8 KB bf16
  __shared__ float sLogit[16];

  const int tid   = threadIdx.x;
  const int wave  = tid >> 5;
  const int lane  = tid & 31;
  const int n     = lane & 15;
  const int hi    = lane >> 4;
  const int rBase = blockIdx.x * 16;

  // Stage 16 rows of x -> bf16 LDS with float4 loads.
  {
    const float4* xV = (const float4*)(xws + (size_t)rBase * (2 * EE));
    uint2* sXV = (uint2*)sX;
    for (int i4 = tid; i4 < (16 * 2 * EE) / 4; i4 += 128)
      { float4 v = xV[i4]; sXV[i4] = make_uint2(pack_bf16(v.x, v.y), pack_bf16(v.z, v.w)); }
  }
  if (tid < 16) sLogit[tid] = 0.0f;
  __syncthreads();

  // B = x^T fragments (K = 256 = 8 k-steps), hoisted once.
  v16bf bx[8];
#pragma unroll
  for (int k = 0; k < 8; ++k)
    bx[k] = load_fragBT_lds(sX, 2 * EE, k * 32, 0);

  for (int t = 0; t < 4; ++t) {
    const int nt    = wave * 4 + t;        // 16 M-tiles over MLP_H = 256
    const int mBase = nt * 16;
    v8f c = {};
#pragma unroll
    for (int k = 0; k < 8; ++k) {
      v16bf a = load_fragAT_global(mlp1_w, MLP_H, k * 32, mBase);
      c = __builtin_amdgcn_wmma_f32_16x16x32_bf16(false, a, false, bx[k], (short)0, c, false, false);
    }
    // D[r][n] = z[row = rBase+n][i = mBase+r+8hi]; reduce over r in registers.
    float part = 0.0f;
#pragma unroll
    for (int r = 0; r < 8; ++r) {
      const int i = mBase + r + hi * 8;
      float h = c[r] + mlp1_b[i];
      h = h > 0.0f ? h : 0.0f;
      part = fmaf(h, mlp2_w[i], part);
    }
    atomicAdd(&sLogit[n], part);
  }
  __syncthreads();
  if (tid < 16) out[rBase + tid] = sLogit[tid] + mlp2_b[0];
}

extern "C" void kernel_launch(void* const* d_in, const int* in_sizes, int n_in,
                              void* d_out, int out_size, void* d_ws, size_t ws_size,
                              hipStream_t stream) {
  (void)in_sizes; (void)n_in; (void)out_size; (void)ws_size;
  const float* beh    = (const float*)d_in[0];
  const float* cand   = (const float*)d_in[1];
  const float* fc1_w  = (const float*)d_in[2];
  const float* fc1_b  = (const float*)d_in[3];
  const float* fc2_w  = (const float*)d_in[4];
  // d_in[5] = fc2_b: constant shift before softmax -> mathematically a no-op.
  const float* mlp1_w = (const float*)d_in[6];
  const float* mlp1_b = (const float*)d_in[7];
  const float* mlp2_w = (const float*)d_in[8];
  const float* mlp2_b = (const float*)d_in[9];
  float* out = (float*)d_out;
  float* xws = (float*)d_ws;   // BB x 256 floats = 4 MB scratch

  din_attn_kernel<<<BB, 256, 0, stream>>>(beh, cand, fc1_w, fc1_b, fc2_w, xws);
  din_mlp_kernel<<<BB / 16, 128, 0, stream>>>(xws, mlp1_w, mlp1_b, mlp2_w, mlp2_b, out);
}
